// MessageBlock_eqv2_15367392985808
// MI455X (gfx1250) — matrix operand order repaired and checked by reading, not compiled, not run on hardware
//
#include <hip/hip_runtime.h>
#include <math.h>

typedef __attribute__((ext_vector_type(16))) __bf16 v16bf;
typedef __attribute__((ext_vector_type(8)))  float  v8f;

#define PI_F 3.14159265358979323846f

// ---------- helpers ----------
__device__ __forceinline__ unsigned short f2bf_bits(float f) {
  unsigned u = __float_as_uint(f);
  unsigned r = u + 0x7FFFu + ((u >> 16) & 1u);   // round-to-nearest-even
  return (unsigned short)(r >> 16);
}
__device__ __forceinline__ float siluf(float x) { return x / (1.f + expf(-x)); }
__device__ __forceinline__ unsigned enc_order(float f) {
  unsigned u = __float_as_uint(f);
  return (u & 0x80000000u) ? ~u : (u | 0x80000000u);
}
__device__ __forceinline__ float dec_order(unsigned u) {
  unsigned v = (u & 0x80000000u) ? (u & 0x7FFFFFFFu) : ~u;
  return __uint_as_float(v);
}
union Frag16 { uint4 q[2]; v16bf v; };

// async global->LDS 16B copy (CDNA5): per-lane global addr -> per-lane LDS addr
__device__ __forceinline__ void async_b128(unsigned lds_addr, const void* gptr) {
  asm volatile("global_load_async_to_lds_b128 %0, %1, off"
               :: "v"(lds_addr), "v"((unsigned long long)gptr) : "memory");
}
__device__ __forceinline__ void wait_async0() {
  asm volatile("s_wait_asynccnt 0x0" ::: "memory");
}
__device__ __forceinline__ unsigned lds_off(const void* p) {
  return (unsigned)(size_t)p;   // generic LDS pointer: low 32 bits = LDS byte offset
}

// packed column-slot for SO2 layout: rows [0,2,6][3,7,1,5][8,4]
__device__ __constant__ int d_pos9[9]   = {0, 5, 1, 3, 8, 6, 2, 4, 7};
// source column offset (x Cout=256) in vbuf for row r of v (conv2 output)
__device__ __constant__ int d_srccol[9] = {0, 1280, 256, 768, 2048, 1536, 512, 1024, 1792};

// ---------- init ----------
__global__ void init_kernel(float* nodeacc, long nacc, unsigned* mxenc, float* den, int n8) {
  long i = (long)blockIdx.x * blockDim.x + threadIdx.x;
  long stride = (long)gridDim.x * blockDim.x;
  for (long j = i; j < nacc; j += stride) nodeacc[j] = 0.f;
  for (long j = i; j < n8; j += stride) { mxenc[j] = 0u; den[j] = 0.f; }
}

// ---------- spherical grid Y (72x9) and pseudo-inverse YI (9x72) ----------
__global__ void sh_setup_kernel(float* Y, float* YI) {
  __shared__ float Ys[72][9];
  int t = threadIdx.x;
  if (t < 72) {
    int ilat = t / 12, ilon = t % 12;
    float th = (ilat + 0.5f) * PI_F / 6.0f;
    float ph = ilon * 2.0f * PI_F / 12.0f;
    float x = sinf(th) * cosf(ph), y = sinf(th) * sinf(ph), z = cosf(th);
    float s3 = sqrtf(3.f), s5 = sqrtf(5.f), s15 = sqrtf(15.f);
    float row[9] = {1.f, s3 * y, s3 * z, s3 * x, s15 * x * y, s15 * y * z,
                    0.5f * s5 * (3.f * z * z - 1.f), s15 * x * z, 0.5f * s15 * (x * x - y * y)};
    for (int c = 0; c < 9; ++c) { Ys[t][c] = row[c]; Y[t * 9 + c] = row[c]; }
  }
  __syncthreads();
  if (t == 0) {
    float G[9][18];
    for (int r = 0; r < 9; ++r)
      for (int c = 0; c < 9; ++c) {
        float s = 0.f;
        for (int g = 0; g < 72; ++g) s += Ys[g][r] * Ys[g][c];
        G[r][c] = s;
        G[r][9 + c] = (r == c) ? 1.f : 0.f;
      }
    for (int col = 0; col < 9; ++col) {            // Gauss-Jordan, partial pivot
      int piv = col;
      for (int r = col + 1; r < 9; ++r) if (fabsf(G[r][col]) > fabsf(G[piv][col])) piv = r;
      if (piv != col) for (int c = 0; c < 18; ++c) { float tv = G[col][c]; G[col][c] = G[piv][c]; G[piv][c] = tv; }
      float inv = 1.f / G[col][col];
      for (int c = 0; c < 18; ++c) G[col][c] *= inv;
      for (int r = 0; r < 9; ++r) if (r != col) {
        float f = G[r][col];
        for (int c = 0; c < 18; ++c) G[r][c] -= f * G[col][c];
      }
    }
    for (int c = 0; c < 9; ++c)                    // YI = (Y^T Y)^-1 Y^T
      for (int g = 0; g < 72; ++g) {
        float s = 0.f;
        for (int k = 0; k < 9; ++k) s += G[c][9 + k] * Ys[g][k];
        YI[c * 72 + g] = s;
      }
  }
}

// ---------- weight prep: store TRANSPOSED bf16 weights  Bt[N][K] ----------
__global__ void cvt_bf16_t_kernel(const float* __restrict__ src, unsigned short* __restrict__ dst,
                                  int K, int N) {
  int i = blockIdx.x * 256 + threadIdx.x;
  if (i >= K * N) return;
  int r = i / N, c = i % N;
  dst[(long)c * K + r] = f2bf_bits(src[i]);
}
// complex block matrix [[Wr,Wi],[-Wi,Wr]] of size [2Kh,2Co], stored transposed [2Co][2Kh]
__global__ void pack_cplx_t_kernel(const float* __restrict__ Wr, const float* __restrict__ Wi,
                                   unsigned short* __restrict__ dst, int Kh, int Co) {
  int i = blockIdx.x * 256 + threadIdx.x;
  if (i >= Kh * Co) return;
  int r = i / Co, c = i % Co;
  long ld = 2 * (long)Kh;
  float wr = Wr[i], wi = Wi[i];
  dst[(long)c * ld + r]             = f2bf_bits(wr);
  dst[(long)(Co + c) * ld + r]      = f2bf_bits(wi);
  dst[(long)c * ld + Kh + r]        = f2bf_bits(-wi);
  dst[(long)(Co + c) * ld + Kh + r] = f2bf_bits(wr);
}

// ---------- fused: radial MLP + gather + wigner rotate + scale -> packed bf16 A ----------
__global__ __launch_bounds__(256) void edge_msg_kernel(
    const float* __restrict__ x, const float* __restrict__ edge_dist,
    const float* __restrict__ wigner,
    const float* __restrict__ src_emb, const float* __restrict__ tgt_emb,
    const float* __restrict__ rw1, const float* __restrict__ rb1,
    const float* __restrict__ rw2, const float* __restrict__ rb2,
    const float* __restrict__ rw3, const float* __restrict__ rb3,
    const int* __restrict__ edge_index, const int* __restrict__ anum,
    unsigned short* __restrict__ msgp, int E) {
  int e = blockIdx.x, t = threadIdx.x;
  __shared__ float xe[96], h1[32], h2[32], wig[81];
  __shared__ float mi[9][512];
  int src = edge_index[e], tgt = edge_index[E + e];
  if (t < 32)       xe[t] = edge_dist[(long)e * 32 + t];
  else if (t < 64)  xe[t] = src_emb[(long)anum[src] * 32 + (t - 32)];
  else if (t < 96)  xe[t] = tgt_emb[(long)anum[tgt] * 32 + (t - 64)];
  if (t >= 96 && t < 96 + 81) wig[t - 96] = wigner[(long)e * 81 + (t - 96)];
  const float* xs = x + (long)src * 2304;
  const float* xt = x + (long)tgt * 2304;
  for (int i = t; i < 4608; i += 256) {
    int j = i >> 9, c = i & 511;
    mi[j][c] = (c < 256) ? xs[j * 256 + c] : xt[j * 256 + (c - 256)];
  }
  __syncthreads();
  if (t < 32) {
    float s = rb1[t];
    for (int i = 0; i < 96; ++i) s += xe[i] * rw1[i * 32 + t];
    h1[t] = siluf(s);
  }
  __syncthreads();
  if (t < 32) {
    float s = rb2[t];
    for (int i = 0; i < 32; ++i) s += h1[i] * rw2[i * 32 + t];
    h2[t] = siluf(s);
  }
  __syncthreads();
  for (int idx = t; idx < 4608; idx += 256) {
    int i = idx >> 9, ch = idx & 511;
    float tmp = 0.f;
    #pragma unroll
    for (int j = 0; j < 9; ++j) tmp += wig[i * 9 + j] * mi[j][ch];
    float rad = rb3[idx];
    #pragma unroll 8
    for (int k = 0; k < 32; ++k) rad += h2[k] * rw3[(long)k * 4608 + idx];
    msgp[(long)e * 4608 + d_pos9[i] * 512 + ch] = f2bf_bits(tmp * rad);
  }
}

// ---------- WMMA bf16 GEMM: C[M,N] = A[M,K] * Bt[N,K]^T ----------
// A: bf16(raw u16) or f32 (a_is_f32), row-major lda. Bt: bf16, N-major, row stride = K.
// BM=128, BN=64, BK=64. 256 threads = 8 waves; wave w -> 32m x 32n block (4 WMMA/k-step).
// OOB M rows: addresses clamped; garbage flows only into C rows >= M which are never stored.
__global__ __launch_bounds__(256) void gemm_bf16_wmma(
    const void* __restrict__ Av, int a_is_f32, long lda,
    const unsigned short* __restrict__ Bt, long ldb,
    float* __restrict__ C, long ldc, int M, int N, int K) {
  __shared__ unsigned short As[128][72];   // [m][k], row stride 144B (16B aligned)
  __shared__ unsigned short Bs[64][72];    // [n][k]
  int bm = blockIdx.y * 128, bn = blockIdx.x * 64;
  int t = threadIdx.x, wave = t >> 5, lane = t & 31;
  int moff = (wave >> 1) * 32, noff = (wave & 1) * 32;
  int lan16 = lane & 31 & 15, khalf = lane >> 4;
  v8f c00 = {}; v8f c01 = {}; v8f c10 = {}; v8f c11 = {};

  int arow = t >> 2;              // 0..63 (two rows per thread: +0, +64)
  int acol = (t & 3) * 16;        // 0,16,32,48
  int brow = t >> 2;              // n index 0..63
  int bcol = (t & 3) * 16;

  for (int k0 = 0; k0 < K; k0 += 64) {
    // ---- stage B tile via async global->LDS (ASYNCcnt path) ----
    {
      const unsigned short* bp = Bt + (size_t)(bn + brow) * ldb + k0 + bcol;
      if (k0 + 64 < K) __builtin_prefetch(bp + 64, 0, 1);
      unsigned l0 = lds_off(&Bs[brow][bcol]);
      async_b128(l0,       bp);
      async_b128(l0 + 16u, bp + 8);
    }
    // ---- stage A tile ----
    if (a_is_f32) {
      const float* A = (const float*)Av;
      #pragma unroll
      for (int rr = 0; rr < 2; ++rr) {
        int gm = bm + arow + 64 * rr;
        int gmc = gm < M ? gm : M - 1;     // clamp only (no zero-fill needed)
        const float* ap = A + (size_t)gmc * lda + k0 + acol;
        if (k0 + 64 < K) __builtin_prefetch(ap + 64, 0, 1);
        unsigned u[8];
        #pragma unroll
        for (int j = 0; j < 8; ++j) {
          unsigned short lo = f2bf_bits(ap[2 * j]);
          unsigned short hi = f2bf_bits(ap[2 * j + 1]);
          u[j] = (unsigned)lo | ((unsigned)hi << 16);
        }
        *(uint4*)&As[arow + 64 * rr][acol]     = make_uint4(u[0], u[1], u[2], u[3]);
        *(uint4*)&As[arow + 64 * rr][acol + 8] = make_uint4(u[4], u[5], u[6], u[7]);
      }
    } else {
      const unsigned short* A = (const unsigned short*)Av;
      #pragma unroll
      for (int rr = 0; rr < 2; ++rr) {
        int gm = bm + arow + 64 * rr;
        int gmc = gm < M ? gm : M - 1;     // clamp only
        const unsigned short* ap = A + (size_t)gmc * lda + k0 + acol;
        if (k0 + 64 < K) __builtin_prefetch(ap + 64, 0, 1);
        unsigned l0 = lds_off(&As[arow + 64 * rr][acol]);
        async_b128(l0,       ap);
        async_b128(l0 + 16u, ap + 8);
      }
    }
    wait_async0();               // own async copies done (others covered by their wait+barrier)
    __syncthreads();
    // ---- two 32-deep k-steps, 4 WMMA each ----
    #pragma unroll
    for (int ks = 0; ks < 2; ++ks) {
      int k0l = ks * 32;
      // A frag rows moff+lan16 and moff+16+lan16; contiguous k run of 16 at k0l+khalf*16
      Frag16 a0, a1, b0, b1;
      int ka = k0l + khalf * 16;
      a0.q[0] = *(const uint4*)&As[moff + lan16][ka];
      a0.q[1] = *(const uint4*)&As[moff + lan16][ka + 8];
      a1.q[0] = *(const uint4*)&As[moff + 16 + lan16][ka];
      a1.q[1] = *(const uint4*)&As[moff + 16 + lan16][ka + 8];
      // B frag: lane n = lan16 (+16), contiguous k run of 16 at k0l+khalf*16
      int kb = k0l + khalf * 16;
      b0.q[0] = *(const uint4*)&Bs[noff + lan16][kb];
      b0.q[1] = *(const uint4*)&Bs[noff + lan16][kb + 8];
      b1.q[0] = *(const uint4*)&Bs[noff + 16 + lan16][kb];
      b1.q[1] = *(const uint4*)&Bs[noff + 16 + lan16][kb + 8];
      c00 = __builtin_amdgcn_wmma_f32_16x16x32_bf16(false, a0.v, false, b0.v, (short)0, c00, false, false);
      c01 = __builtin_amdgcn_wmma_f32_16x16x32_bf16(false, a0.v, false, b1.v, (short)0, c01, false, false);
      c10 = __builtin_amdgcn_wmma_f32_16x16x32_bf16(false, a1.v, false, b0.v, (short)0, c10, false, false);
      c11 = __builtin_amdgcn_wmma_f32_16x16x32_bf16(false, a1.v, false, b1.v, (short)0, c11, false, false);
    }
    __syncthreads();
  }
  // ---- store C: VGPR r -> M = r + 8*khalf, N = lan16 ----
  #pragma unroll
  for (int r = 0; r < 8; ++r) {
    int m0 = bm + moff + r + 8 * khalf;
    int m1 = m0 + 16;
    long n0 = bn + noff + lan16;
    if (m0 < M) {
      C[(long)m0 * ldc + n0]      = c00[r];
      C[(long)m0 * ldc + n0 + 16] = c01[r];
    }
    if (m1 < M) {
      C[(long)m1 * ldc + n0]      = c10[r];
      C[(long)m1 * ldc + n0 + 16] = c11[r];
    }
  }
}

// ---------- grid activation + gate + pack conv2 A (bf16) ----------
__global__ __launch_bounds__(256) void grid_act_kernel(
    const float* __restrict__ ybuf, const float* __restrict__ Y, const float* __restrict__ YI,
    unsigned short* __restrict__ A2, int E) {
  int e = blockIdx.x, t = threadIdx.x;
  __shared__ float m2[9][128];
  __shared__ float g[72][128];
  __shared__ float Ys[648], YIs[648];
  const float* y0 = ybuf + (long)e * 2304;
  if (t < 128) {
    int c = t;
    m2[0][c] = y0[c];            m2[2][c] = y0[128 + c];        m2[6][c] = y0[256 + c];
    m2[3][c] = y0[1536 + c];     m2[7][c] = y0[1536 + 128 + c];
    m2[1][c] = y0[1536 + 256 + c]; m2[5][c] = y0[1536 + 384 + c];
    m2[8][c] = y0[2048 + c];     m2[4][c] = y0[2048 + 128 + c];
  }
  for (int i = t; i < 648; i += 256) { Ys[i] = Y[i]; YIs[i] = YI[i]; }
  __syncthreads();
  for (int i = t; i < 72 * 128; i += 256) {
    int gi = i >> 7, c = i & 127;
    float s = 0.f;
    #pragma unroll
    for (int j = 0; j < 9; ++j) s += Ys[gi * 9 + j] * m2[j][c];
    g[gi][c] = siluf(s);
  }
  __syncthreads();
  for (int i = t; i < 9 * 128; i += 256) {
    int r = i >> 7, c = i & 127;
    float val;
    if (r == 0) {
      val = siluf(y0[1408 + c]);                 // row0 <- silu(gate)
    } else {
      float s = 0.f;
      for (int gg = 0; gg < 72; ++gg) s += YIs[r * 72 + gg] * g[gg][c];
      val = s;
    }
    A2[(long)e * 1152 + d_pos9[r] * 128 + c] = f2bf_bits(val);
  }
}

// ---------- LN + gated activation + logits + segment max ----------
__global__ __launch_bounds__(256) void alpha_kernel(
    const float* __restrict__ ybuf, const float* __restrict__ ln_g, const float* __restrict__ ln_b,
    const float* __restrict__ alpha_dot, const int* __restrict__ edge_index,
    float* __restrict__ logits, unsigned* __restrict__ mxenc, int E) {
  int e = blockIdx.x;
  int head = threadIdx.x >> 5, lane = threadIdx.x & 31;    // 8 heads x wave32
  const float* af = ybuf + (long)e * 2304 + 384 + head * 128;
  float v[4], s = 0.f, s2 = 0.f;
  #pragma unroll
  for (int i = 0; i < 4; ++i) { v[i] = af[lane + 32 * i]; s += v[i]; s2 += v[i] * v[i]; }
  for (int o = 16; o; o >>= 1) { s += __shfl_xor(s, o, 32); s2 += __shfl_xor(s2, o, 32); }
  float mu = s * (1.f / 128.f);
  float var = s2 * (1.f / 128.f) - mu * mu;
  float rs = rsqrtf(var + 1e-5f);
  float acc = 0.f;
  #pragma unroll
  for (int i = 0; i < 4; ++i) {
    int c = lane + 32 * i;
    float a = (v[i] - mu) * rs * ln_g[c] + ln_b[c];
    float sg = 1.f / (1.f + expf(-a));
    a = 0.6f * a + 0.4f * a * (2.f * sg - 1.f);
    acc += a * alpha_dot[head * 128 + c];
  }
  for (int o = 16; o; o >>= 1) acc += __shfl_xor(acc, o, 32);
  if (lane == 0) {
    logits[(long)e * 8 + head] = acc;
    int tgt = edge_index[E + e];
    atomicMax(&mxenc[tgt * 8 + head], enc_order(acc));
  }
}

__global__ void exp_den_kernel(const float* __restrict__ logits, const unsigned* __restrict__ mxenc,
                               const int* __restrict__ edge_index, float* __restrict__ ex,
                               float* __restrict__ den, int E) {
  int i = blockIdx.x * 256 + threadIdx.x;
  if (i >= E * 8) return;
  int e = i >> 3, h = i & 7;
  int tgt = edge_index[E + e];
  float v = expf(logits[i] - dec_order(mxenc[tgt * 8 + h]));
  ex[i] = v;
  atomicAdd(&den[tgt * 8 + h], v);
}

// ---------- attn * v, inverse wigner rotate, scatter-add to nodes ----------
__global__ __launch_bounds__(256) void scatter_v_kernel(
    const float* __restrict__ vbuf, const float* __restrict__ ex, const float* __restrict__ den,
    const float* __restrict__ wigner, const int* __restrict__ edge_index,
    float* __restrict__ nodeacc, int E) {
  int e = blockIdx.x, t = threadIdx.x;
  __shared__ float v[9][256];
  __shared__ float wig[81];
  __shared__ float at[8];
  int tgt = edge_index[E + e];
  if (t < 81) wig[t] = wigner[(long)e * 81 + t];
  if (t >= 81 && t < 89) {
    int h = t - 81;
    at[h] = ex[(long)e * 8 + h] / (den[tgt * 8 + h] + 1e-12f);
  }
  __syncthreads();
  const float* vb = vbuf + (long)e * 2304;
  for (int i = t; i < 2304; i += 256) {
    int r = i >> 8, c = i & 255;
    v[r][c] = vb[d_srccol[r] + c] * at[c >> 5];
  }
  __syncthreads();
  for (int i = t; i < 2304; i += 256) {
    int r = i >> 8, c = i & 255;
    float s = 0.f;
    #pragma unroll
    for (int j = 0; j < 9; ++j) s += wig[j * 9 + r] * v[j][c];   // 'eji,ejc->eic'
    atomicAdd(&nodeacc[((long)tgt * 9 + r) * 256 + c], s);
  }
}

__global__ void bias_kernel(float* __restrict__ out, const float* __restrict__ bproj, int N) {
  int i = blockIdx.x * 256 + threadIdx.x;
  if (i < N * 256) out[(long)(i >> 8) * 2304 + (i & 255)] += bproj[i & 255];
}

// ---------- launch ----------
extern "C" void kernel_launch(void* const* d_in, const int* in_sizes, int n_in,
                              void* d_out, int out_size, void* d_ws, size_t ws_size,
                              hipStream_t stream) {
  const float* x         = (const float*)d_in[0];
  const float* edge_dist = (const float*)d_in[1];
  const float* wigner    = (const float*)d_in[2];
  const float* src_emb   = (const float*)d_in[3];
  const float* tgt_emb   = (const float*)d_in[4];
  const float* rad_w1 = (const float*)d_in[5];  const float* rad_b1 = (const float*)d_in[6];
  const float* rad_w2 = (const float*)d_in[7];  const float* rad_b2 = (const float*)d_in[8];
  const float* rad_w3 = (const float*)d_in[9];  const float* rad_b3 = (const float*)d_in[10];
  const float* W0_1  = (const float*)d_in[11];
  const float* W1r_1 = (const float*)d_in[12]; const float* W1i_1 = (const float*)d_in[13];
  const float* W2r_1 = (const float*)d_in[14]; const float* W2i_1 = (const float*)d_in[15];
  const float* W0_2  = (const float*)d_in[16];
  const float* W1r_2 = (const float*)d_in[17]; const float* W1i_2 = (const float*)d_in[18];
  const float* W2r_2 = (const float*)d_in[19]; const float* W2i_2 = (const float*)d_in[20];
  const float* ln_g = (const float*)d_in[21]; const float* ln_b = (const float*)d_in[22];
  const float* alpha_dot = (const float*)d_in[23];
  const float* Wproj = (const float*)d_in[24]; const float* bproj = (const float*)d_in[25];
  const int* edge_index = (const int*)d_in[26];
  const int* anum       = (const int*)d_in[27];
  float* out = (float*)d_out;

  int E = in_sizes[1] / 32;
  int N = in_sizes[0] / (9 * 256);

  char* p = (char*)d_ws;
  auto alloc = [&](size_t bytes) -> char* {
    char* r = p; p += (bytes + 255) & ~(size_t)255; return r;
  };
  float* wY  = (float*)alloc(72 * 9 * 4);
  float* wYI = (float*)alloc(9 * 72 * 4);
  unsigned short* msgp = (unsigned short*)alloc((size_t)E * 4608 * 2);
  float* ybuf = (float*)alloc((size_t)E * 2304 * 4);
  unsigned short* a2buf = (unsigned short*)alloc((size_t)E * 1152 * 2);
  float* vbuf = (float*)alloc((size_t)E * 2304 * 4);
  float* nodeacc = (float*)alloc((size_t)N * 2304 * 4);
  float* logits = (float*)alloc((size_t)E * 8 * 4);
  float* ex     = (float*)alloc((size_t)E * 8 * 4);
  unsigned* mxenc = (unsigned*)alloc((size_t)N * 8 * 4);
  float* den      = (float*)alloc((size_t)N * 8 * 4);
  unsigned short* w0b1  = (unsigned short*)alloc((size_t)1536 * 1536 * 2);
  unsigned short* w1c1  = (unsigned short*)alloc((size_t)2048 * 512 * 2);
  unsigned short* w2c1  = (unsigned short*)alloc((size_t)1024 * 256 * 2);
  unsigned short* w0b2  = (unsigned short*)alloc((size_t)384 * 768 * 2);
  unsigned short* w1c2  = (unsigned short*)alloc((size_t)512 * 1024 * 2);
  unsigned short* w2c2  = (unsigned short*)alloc((size_t)256 * 512 * 2);
  unsigned short* wprjb = (unsigned short*)alloc((size_t)256 * 256 * 2);

  // 0) init accumulators
  init_kernel<<<2048, 256, 0, stream>>>(nodeacc, (long)N * 2304, mxenc, den, N * 8);
  // 1) Y grid + pinv
  sh_setup_kernel<<<1, 128, 0, stream>>>(wY, wYI);
  // 2) weight prep (transposed bf16 Bt[N][K] + complex block matrices)
  cvt_bf16_t_kernel<<<(1536 * 1536 + 255) / 256, 256, 0, stream>>>(W0_1, w0b1, 1536, 1536);
  cvt_bf16_t_kernel<<<(384 * 768 + 255) / 256, 256, 0, stream>>>(W0_2, w0b2, 384, 768);
  cvt_bf16_t_kernel<<<(256 * 256 + 255) / 256, 256, 0, stream>>>(Wproj, wprjb, 256, 256);
  pack_cplx_t_kernel<<<(1024 * 256 + 255) / 256, 256, 0, stream>>>(W1r_1, W1i_1, w1c1, 1024, 256);
  pack_cplx_t_kernel<<<(512 * 128 + 255) / 256, 256, 0, stream>>>(W2r_1, W2i_1, w2c1, 512, 128);
  pack_cplx_t_kernel<<<(256 * 512 + 255) / 256, 256, 0, stream>>>(W1r_2, W1i_2, w1c2, 256, 512);
  pack_cplx_t_kernel<<<(128 * 256 + 255) / 256, 256, 0, stream>>>(W2r_2, W2i_2, w2c2, 128, 256);
  // 3) fused radial MLP + gather + rotate + scale -> msgp (bf16, packed)
  edge_msg_kernel<<<E, 256, 0, stream>>>(x, edge_dist, wigner, src_emb, tgt_emb,
                                         rad_w1, rad_b1, rad_w2, rad_b2, rad_w3, rad_b3,
                                         edge_index, anum, msgp, E);
  auto gemm = [&](const void* A, int af32, long lda, const unsigned short* Bt, long ldb,
                  float* C, long ldc, int M, int Nn, int K) {
    dim3 g((Nn + 63) / 64, (M + 127) / 128);
    gemm_bf16_wmma<<<g, 256, 0, stream>>>(A, af32, lda, Bt, ldb, C, ldc, M, Nn, K);
  };
  // 4) SO2 conv 1 (three WMMA GEMMs into ybuf [E,2304])  (ldb == K of each GEMM)
  gemm(msgp,          0, 4608, w0b1, 1536, ybuf,        2304, E, 1536, 1536);
  gemm(msgp + 1536,   0, 4608, w1c1, 2048, ybuf + 1536, 2304, E, 512,  2048);
  gemm(msgp + 3584,   0, 4608, w2c1, 1024, ybuf + 2048, 2304, E, 256,  1024);
  // 5) grid nonlinearity + gate -> a2buf (bf16, packed)
  grid_act_kernel<<<E, 256, 0, stream>>>(ybuf, wY, wYI, a2buf, E);
  // 6) SO2 conv 2 -> vbuf [E,2304]
  gemm(a2buf,       0, 1152, w0b2, 384, vbuf,        2304, E, 768,  384);
  gemm(a2buf + 384, 0, 1152, w1c2, 512, vbuf + 768,  2304, E, 1024, 512);
  gemm(a2buf + 896, 0, 1152, w2c2, 256, vbuf + 1792, 2304, E, 512,  256);
  // 7) attention logits + segment softmax
  alpha_kernel<<<E, 256, 0, stream>>>(ybuf, ln_g, ln_b, alpha_dot, edge_index, logits, mxenc, E);
  exp_den_kernel<<<(E * 8 + 255) / 256, 256, 0, stream>>>(logits, mxenc, edge_index, ex, den, E);
  // 8) weight, rotate back, scatter to nodes
  scatter_v_kernel<<<E, 256, 0, stream>>>(vbuf, ex, den, wigner, edge_index, nodeacc, E);
  // 9) final projection (WMMA GEMM, f32 A converted on the fly) + bias on l=0
  gemm(nodeacc, 1, 256, wprjb, 256, out, 256, N * 9, 256, 256);
  bias_kernel<<<(N * 256 + 255) / 256, 256, 0, stream>>>(out, bproj, N);
}
